// GCN_30374008717350
// MI455X (gfx1250) — matrix operand order repaired and checked by reading, not compiled, not run on hardware
//
#include <hip/hip_runtime.h>
#include <hip/hip_bf16.h>
#include <math.h>

typedef __attribute__((ext_vector_type(16))) _Float16 v16h;
typedef __attribute__((ext_vector_type(8)))  float    v8f;

// ---------------------------------------------------------------------------
// Pack a row-major f32 weight [K x Ncols] into per-lane B-fragments for
// v_wmma_f32_16x16x32_f16.  Layout: frag[(nt*ktiles+kt)*32 + lane][i] =
// W[kt*32 + (lane>>4)*16 + i][nt*16 + (lane&15)]  (zero-padded for n>=Ncols).
// ---------------------------------------------------------------------------
__global__ __launch_bounds__(32)
void pack_B(const float* __restrict__ W, _Float16* __restrict__ Bp,
            int Ncols, int ktiles) {
  int nt = blockIdx.x, kt = blockIdx.y;
  int lane = threadIdx.x;
  int n = nt * 16 + (lane & 15);
  int kbase = kt * 32 + (lane >> 4) * 16;
  _Float16* dst = Bp + ((size_t)(nt * ktiles + kt) * 32 + lane) * 16;
#pragma unroll
  for (int i = 0; i < 16; ++i) {
    float v = (n < Ncols) ? W[(size_t)(kbase + i) * Ncols + n] : 0.0f;
    dst[i] = (_Float16)v;
  }
}

__global__ void f32_to_f16(const float* __restrict__ src,
                           _Float16* __restrict__ dst, size_t n) {
  size_t i = (size_t)blockIdx.x * blockDim.x + threadIdx.x;
  if (i < n) dst[i] = (_Float16)src[i];
}

// ---------------------------------------------------------------------------
// One wave (32 threads) computes one 16x16 D tile, looping over K in steps
// of 32 with v_wmma_f32_16x16x32_f16.  Epilogue: optional bias, ReLU,
// f32 store (stride ldF) and/or f16 store (stride ldH).
// ---------------------------------------------------------------------------
__global__ __launch_bounds__(32)
void wmma_gemm(const _Float16* __restrict__ A, const _Float16* __restrict__ Bp,
               const float* __restrict__ bias,
               float* __restrict__ outF, _Float16* __restrict__ outH,
               int M, int K, int Ncols, int ldF, int ldH, int relu) {
  int mt = blockIdx.x;
  int nt = blockIdx.y;
  int ktiles = K >> 5;
  int lane = threadIdx.x;
  int hid = lane >> 4;            // half-wave id
  int m0 = mt * 16;
  int mrow = m0 + (lane & 15);
  if (mrow >= M) mrow = M - 1;    // clamp (no EXEC divergence before WMMA)

  v8f c = {0.f, 0.f, 0.f, 0.f, 0.f, 0.f, 0.f, 0.f};
  const _Float16* bfrag =
      Bp + ((size_t)nt * ktiles * 32 + lane) * 16;

  for (int kt = 0; kt < ktiles; ++kt) {
    // A fragment: contiguous K order, element s -> K = 16h + s (matches pack_B)
    union { v16h v; uint4 q[2]; } a;
    const uint4* ap4 = (const uint4*)(A + (size_t)mrow * K + kt * 32);
    a.q[0] = ap4[2 * hid];        // K = 16h .. 16h+7
    a.q[1] = ap4[2 * hid + 1];    // K = 16h+8 .. 16h+15

    v16h b = *(const v16h*)(bfrag + (size_t)kt * 32 * 16);

    c = __builtin_amdgcn_wmma_f32_16x16x32_f16(
        /*neg_a=*/false, a.v, /*neg_b=*/false, b,
        /*c_mod=*/(short)0, c, /*reuse_a=*/false, /*reuse_b=*/false);
  }

  int n = nt * 16 + (lane & 15);
  float bv = (bias != nullptr && n < Ncols) ? bias[n] : 0.0f;
#pragma unroll
  for (int r = 0; r < 8; ++r) {
    int m = m0 + hid * 8 + r;   // C/D layout: lanes 16-31 hold M = 8..15
    if (m < M && n < Ncols) {
      float v = c[r] + bv;
      if (relu) v = fmaxf(v, 0.0f);
      if (outF) outF[(size_t)m * ldF + n] = v;
      if (outH) outH[(size_t)m * ldH + n] = (_Float16)v;
    }
  }
}

// ---------------------------------------------------------------------------
// Degree / normalization helpers
// ---------------------------------------------------------------------------
__global__ void deg_init(float* __restrict__ deg, int Nn) {
  int i = blockIdx.x * blockDim.x + threadIdx.x;
  if (i < Nn) deg[i] = 1.0f;                  // self-loop weight
}

__global__ void deg_edges(const long long* __restrict__ ei,
                          const float* __restrict__ ew,
                          float* __restrict__ deg, long long E) {
  long long e = (long long)blockIdx.x * blockDim.x + threadIdx.x;
  if (e < E) atomicAdd(&deg[(size_t)ei[E + e]], ew[e]);  // target = col
}

__global__ void make_dinv(const float* __restrict__ deg,
                          float* __restrict__ dinv, int Nn) {
  int i = blockIdx.x * blockDim.x + threadIdx.x;
  if (i < Nn) dinv[i] = rsqrtf(deg[i]);       // deg >= 1 always
}

// agg[i][f] = dinv[i]^2 * t[i][f]   (self-loop contribution, also zero-inits)
__global__ void self_init(const float* __restrict__ t,
                          const float* __restrict__ dinv,
                          float* __restrict__ agg) {
  int node = blockIdx.x, f = threadIdx.x, Hh = blockDim.x;
  float d = dinv[node];
  size_t idx = (size_t)node * Hh + f;
  agg[idx] = d * d * t[idx];
}

// one block (64 threads) per edge: gather t[row][*], scatter-add to agg[col][*]
__global__ void edge_agg(const long long* __restrict__ ei,
                         const float* __restrict__ ew,
                         const float* __restrict__ dinv,
                         const float* __restrict__ t,
                         float* __restrict__ agg, long long E) {
  long long e = blockIdx.x;
  int f = threadIdx.x, Hh = blockDim.x;
  int r  = (int)ei[e];
  int cl = (int)ei[E + e];
  float norm = dinv[r] * ew[e] * dinv[cl];
  atomicAdd(&agg[(size_t)cl * Hh + f], norm * t[(size_t)r * Hh + f]);
}

// h16[i][f] = f16( relu(agg[i][f] + b[f]) )
__global__ void bias_relu_h16(const float* __restrict__ agg,
                              const float* __restrict__ b,
                              _Float16* __restrict__ h16) {
  int node = blockIdx.x, f = threadIdx.x, Hh = blockDim.x;
  size_t idx = (size_t)node * Hh + f;
  h16[idx] = (_Float16)fmaxf(agg[idx] + b[f], 0.0f);
}

// in-place row log-softmax over C columns
__global__ void log_softmax_rows(float* __restrict__ out, int Nn, int C) {
  int i = blockIdx.x * blockDim.x + threadIdx.x;
  if (i >= Nn) return;
  float* row = out + (size_t)i * C;
  float mx = -INFINITY;
  for (int c = 0; c < C; ++c) mx = fmaxf(mx, row[c]);
  float s = 0.0f;
  for (int c = 0; c < C; ++c) s += __expf(row[c] - mx);
  float ls = __logf(s);
  for (int c = 0; c < C; ++c) row[c] = row[c] - mx - ls;
}

// ---------------------------------------------------------------------------
extern "C" void kernel_launch(void* const* d_in, const int* in_sizes, int n_in,
                              void* d_out, int out_size, void* d_ws, size_t ws_size,
                              hipStream_t stream) {
  const float*     x   = (const float*)d_in[0];
  const long long* ei  = (const long long*)d_in[1];   // int64 edge_index [2,E]
  const float*     ew  = (const float*)d_in[2];
  const float*     Wf  = (const float*)d_in[3];
  const float*     bf  = (const float*)d_in[4];
  const float*     Wc1 = (const float*)d_in[5];
  const float*     bc1 = (const float*)d_in[6];
  const float*     Wc2 = (const float*)d_in[7];
  const float*     bc2 = (const float*)d_in[8];
  const float*     Wo  = (const float*)d_in[9];
  const float*     bo  = (const float*)d_in[10];
  float* out = (float*)d_out;

  const int  H  = in_sizes[4];              // 64
  const int  C  = in_sizes[10];             // 40
  const int  F  = in_sizes[3] / H;          // 128
  const int  Nn = in_sizes[0] / F;          // 100000
  const long long E = in_sizes[2];          // 1600000

  // ---- workspace partition (256B aligned) ----
  char* p = (char*)d_ws;
  auto alloc = [&](size_t bytes) -> char* {
    char* r = p; p += (bytes + 255) & ~(size_t)255; return r;
  };
  float*     deg  = (float*)alloc((size_t)Nn * 4);
  float*     dinv = (float*)alloc((size_t)Nn * 4);
  _Float16*  x16  = (_Float16*)alloc((size_t)Nn * F * 2);
  _Float16*  h16  = (_Float16*)alloc((size_t)Nn * H * 2);
  float*     t    = (float*)alloc((size_t)Nn * H * 4);
  float*     agg  = (float*)alloc((size_t)Nn * H * 4);

  const int ntF = H / 16, ktF = F / 32;           // first lin: 128 -> 64
  const int ntC = H / 16, ktC = H / 32;           // convs:      64 -> 64
  const int ntO = (C + 15) / 16, ktO = H / 32;    // out:        64 -> 40 (pad 48)
  _Float16* BpF = (_Float16*)alloc((size_t)ntF * ktF * 512 * 2);
  _Float16* Bp1 = (_Float16*)alloc((size_t)ntC * ktC * 512 * 2);
  _Float16* Bp2 = (_Float16*)alloc((size_t)ntC * ktC * 512 * 2);
  _Float16* BpO = (_Float16*)alloc((size_t)ntO * ktO * 512 * 2);

  const int mtiles = (Nn + 15) / 16;

  // ---- weight packing + input conversion ----
  pack_B<<<dim3(ntF, ktF), 32, 0, stream>>>(Wf,  BpF, H, ktF);
  pack_B<<<dim3(ntC, ktC), 32, 0, stream>>>(Wc1, Bp1, H, ktC);
  pack_B<<<dim3(ntC, ktC), 32, 0, stream>>>(Wc2, Bp2, H, ktC);
  pack_B<<<dim3(ntO, ktO), 32, 0, stream>>>(Wo,  BpO, C, ktO);
  {
    size_t n = (size_t)Nn * F;
    f32_to_f16<<<(unsigned)((n + 255) / 256), 256, 0, stream>>>(x, x16, n);
  }

  // ---- symmetric normalization ----
  deg_init<<<(Nn + 255) / 256, 256, 0, stream>>>(deg, Nn);
  deg_edges<<<(unsigned)((E + 255) / 256), 256, 0, stream>>>(ei, ew, deg, E);
  make_dinv<<<(Nn + 255) / 256, 256, 0, stream>>>(deg, dinv, Nn);

  // ---- layer 1: h = relu(x @ Wf + bf) -> f16 ----
  wmma_gemm<<<dim3(mtiles, ntF), 32, 0, stream>>>(
      x16, BpF, bf, nullptr, h16, Nn, F, H, 0, H, 1);

  // ---- GCNConv 1 ----
  wmma_gemm<<<dim3(mtiles, ntC), 32, 0, stream>>>(
      h16, Bp1, nullptr, t, nullptr, Nn, H, H, H, 0, 0);
  self_init<<<Nn, H, 0, stream>>>(t, dinv, agg);
  edge_agg<<<(unsigned)E, H, 0, stream>>>(ei, ew, dinv, t, agg, E);
  bias_relu_h16<<<Nn, H, 0, stream>>>(agg, bc1, h16);

  // ---- GCNConv 2 ----
  wmma_gemm<<<dim3(mtiles, ntC), 32, 0, stream>>>(
      h16, Bp2, nullptr, t, nullptr, Nn, H, H, H, 0, 0);
  self_init<<<Nn, H, 0, stream>>>(t, dinv, agg);
  edge_agg<<<(unsigned)E, H, 0, stream>>>(ei, ew, dinv, t, agg, E);
  bias_relu_h16<<<Nn, H, 0, stream>>>(agg, bc2, h16);

  // ---- logits straight into d_out, then in-place log-softmax ----
  wmma_gemm<<<dim3(mtiles, ntO), 32, 0, stream>>>(
      h16, BpO, bo, out, nullptr, Nn, H, C, C, 0, 0);
  log_softmax_rows<<<(Nn + 255) / 256, 256, 0, stream>>>(out, Nn, C);
}
